// SimplifiedMamba_27848567947365
// MI455X (gfx1250) — matrix operand order repaired and checked by reading, not compiled, not run on hardware
//
#include <hip/hip_runtime.h>
#include <hip/hip_bf16.h>
#include <math.h>

// ---------------------------------------------------------------------------
// SimplifiedMamba on MI455X (gfx1250, wave32).
//   heavy GEMMs (X@W_in^T, X@W_gate^T: 430 GFLOP) -> bf16 WMMA, f32 accum
//   M=64 row tile per block: each W-fragment feeds 4 A-fragments (8 WMMAs),
//   cutting L2 weight traffic 4x vs an M=16 tile (25.6 GB -> 6.4 GB).
//   A-strip staged to LDS with GLOBAL_LOAD_ASYNC_TO_LDS_B128 (inline asm).
//   gating + W_B reduction fused into GEMM epilogue (never materialize 419MB)
//   sigmoid uses v_rcp_f32 (result is truncated to bf16 anyway) to avoid the
//   IEEE div expansion (div_scale/div_fmas/div_fixup) seen in the disasm.
// ---------------------------------------------------------------------------

typedef __bf16 v16bf __attribute__((ext_vector_type(16)));
typedef __bf16 v8bf  __attribute__((ext_vector_type(8)));
typedef float  v8f   __attribute__((ext_vector_type(8)));

#define DM      1024      // d_model
#define DI      2048      // d_inner
#define DSZ     16        // d_state
#define SEQLEN  100
#define NBATCH  512
#define NROWS   (NBATCH * SEQLEN)   // 51200
#define MTILE   64                  // rows per block in k_core

#define USE_ASYNC_LDS 1

static __device__ __forceinline__ v16bf cat8(v8bf lo, v8bf hi) {
  return __builtin_shufflevector(lo, hi, 0,1,2,3,4,5,6,7,8,9,10,11,12,13,14,15);
}

static __device__ __forceinline__ float fast_rcp(float x) {
#if __has_builtin(__builtin_amdgcn_rcpf)
  return __builtin_amdgcn_rcpf(x);    // v_rcp_f32, ~1ulp: plenty for bf16
#else
  return 1.0f / x;
#endif
}

#if USE_ASYNC_LDS
// GLOBAL_LOAD_ASYNC_TO_LDS_B128 (GV mode): VDST = per-lane LDS byte offset,
// VADDR = 64-bit global address. Tracked by ASYNCcnt.
static __device__ __forceinline__ void async_ld16(const void* g, void* l) {
  unsigned int loff = (unsigned int)(unsigned long long)l;  // flat->LDS offset
  asm volatile("global_load_async_to_lds_b128 %0, %1, off"
               :: "v"(loff), "v"(g) : "memory");
}
#endif

// ---- 1) x + pos_emb -> bf16 -----------------------------------------------
__global__ __launch_bounds__(256) void k_prep_x(const float* __restrict__ x,
                                                const float* __restrict__ pos,
                                                __bf16* __restrict__ xb) {
  size_t idx = (size_t)blockIdx.x * 256 + threadIdx.x;   // exact grid
  int d = (int)(idx & (DM - 1));
  int s = (int)((idx >> 10) % SEQLEN);
  xb[idx] = (__bf16)(x[idx] + pos[s * DM + d]);
}

// ---- 2) weights -> bf16 ----------------------------------------------------
__global__ __launch_bounds__(256) void k_prep_w(const float* __restrict__ Win,
                                                const float* __restrict__ Wg,
                                                const float* __restrict__ WB,
                                                const float* __restrict__ WC,
                                                __bf16* __restrict__ Winb,
                                                __bf16* __restrict__ Wgb,
                                                __bf16* __restrict__ WBb,
                                                __bf16* __restrict__ WCb) {
  int idx = blockIdx.x * 256 + threadIdx.x;
  const int NW = DI * DM;                 // 2097152
  if (idx < NW)                 Winb[idx]            = (__bf16)Win[idx];
  else if (idx < 2 * NW)        Wgb[idx - NW]        = (__bf16)Wg[idx - NW];
  else if (idx < 2*NW + DSZ*DI) WBb[idx - 2*NW]      = (__bf16)WB[idx - 2*NW];
  else                          WCb[idx - 2*NW - DSZ*DI] = (__bf16)WC[idx - 2*NW - DSZ*DI];
}

// ---- 3) fused: x_proj, gate, x_gated, B = xg@W_B^T + b_B, C = x@W_C^T + b_C
// One block = 64 rows (4 M-tiles). A-strip 64x1024 bf16 in LDS, 2064B row
// stride -> 16-lane A-frag b128 reads hit 16 distinct 4-bank groups.
// ~150KB LDS total: 2 blocks co-resident per 320KB WGP.
__global__ __launch_bounds__(256) void k_core(
    const __bf16* __restrict__ xb,   const __bf16* __restrict__ Winb,
    const __bf16* __restrict__ Wgb,  const __bf16* __restrict__ WBb,
    const __bf16* __restrict__ WCb,  const float* __restrict__ b_in,
    const float* __restrict__ b_gate,const float* __restrict__ b_B,
    const float* __restrict__ b_C,   float* __restrict__ Bm,
    float* __restrict__ Cm)
{
  __shared__ __align__(16) __bf16 Astrip[MTILE][DM + 8];  // 132.1 KB
  __shared__ __align__(16) __bf16 stage[8][16][40];       // 80B row stride
  __shared__ float redB[MTILE * DSZ];                     // 4 KB
  __shared__ float redC[MTILE * DSZ];                     // 4 KB

  const int tid  = threadIdx.x;
  const int wave = tid >> 5;          // 8 waves (wave32)
  const int lane = tid & 31;
  const int l16  = lane & 15;
  const bool hi  = lane >= 16;
  const int row0 = blockIdx.x * MTILE;
  const int abase = hi ? 16 : 0;      // A-frag: lanes>=16 hold K+16 halves

  #pragma unroll
  for (int q = 0; q < 4; ++q) redB[tid + 256 * q] = 0.0f;

  { // stage A strip: 64 rows x 1024 K of bf16 (128 KB)
    const __bf16* gsrc = xb + (size_t)row0 * DM;
#if USE_ASYNC_LDS
    #pragma unroll
    for (int j = 0; j < 32; ++j) {
      int q = tid + 256 * j;          // 8192 16-byte chunks
      int r = q >> 7;                 // 128 chunks per row
      int c = q & 127;
      async_ld16(gsrc + (size_t)r * DM + c * 8, &Astrip[r][c * 8]);
    }
    asm volatile("s_wait_asynccnt 0x0" ::: "memory");
#else
    const unsigned int* xg = (const unsigned int*)gsrc;
    #pragma unroll
    for (int j = 0; j < 128; ++j) {
      int p  = tid + 256 * j;         // 32768 dword pairs
      int r  = p >> 9;
      int kp = p & 511;
      *(unsigned int*)&Astrip[r][kp * 2] = xg[r * 512 + kp];
    }
#endif
  }
  __syncthreads();

  v8f accB[4] = {};

  for (int t = 0; t < 16; ++t) {
    const int n0 = wave * 256 + t * 16;      // this wave's 16 inner cols
    v8f accP[4] = {};
    v8f accG[4] = {};
    // B-operand (KxN): lane holds col n0+l16; lanes>=16 take K 16..31
    const __bf16* wi = Winb + (size_t)(n0 + l16) * DM + (hi ? 16 : 0);
    const __bf16* wg = Wgb  + (size_t)(n0 + l16) * DM + (hi ? 16 : 0);
    #pragma unroll 2
    for (int k0 = 0; k0 < DM; k0 += 32) {
      v16bf bi = cat8(*(const v8bf*)(wi + k0), *(const v8bf*)(wi + k0 + 8));
      v16bf bg = cat8(*(const v8bf*)(wg + k0), *(const v8bf*)(wg + k0 + 8));
      __builtin_prefetch(wi + k0 + 128, 0, 1);
      __builtin_prefetch(wg + k0 + 128, 0, 1);
      #pragma unroll
      for (int mi = 0; mi < 4; ++mi) {   // reuse W-frags across 4 A-frags
        v16bf a = cat8(*(const v8bf*)&Astrip[mi * 16 + l16][k0 + abase],
                       *(const v8bf*)&Astrip[mi * 16 + l16][k0 + abase + 8]);
        accP[mi] = __builtin_amdgcn_wmma_f32_16x16x32_bf16(false, a, false, bi,
                                                 (short)0, accP[mi], false, false);
        accG[mi] = __builtin_amdgcn_wmma_f32_16x16x32_bf16(false, a, false, bg,
                                                 (short)0, accG[mi], false, false);
      }
    }
    // epilogue: bias + sigmoid gate; W_B fragment shared by all 4 M-tiles
    const float bin = b_in[n0 + l16];
    const float bgt = b_gate[n0 + l16];
    v16bf wb = {};
    if (!hi) {  // lanes 0..15 carry K=0..15 of W_B^T chunk; reconverges below
      const __bf16* pb = WBb + (size_t)l16 * DI + n0;
      wb = cat8(*(const v8bf*)pb, *(const v8bf*)(pb + 8));
    }
    v8bf z8 = {};
    #pragma unroll
    for (int mi = 0; mi < 4; ++mi) {
      #pragma unroll
      for (int v = 0; v < 8; ++v) {
        float xp = accP[mi][v] + bin;
        float gt = accG[mi][v] + bgt;
        float sg = fast_rcp(1.0f + __expf(-gt));   // v_exp + v_rcp only
        stage[wave][v + (hi ? 8 : 0)][l16] = (__bf16)(xp * sg);  // C->A relayout
      }
      asm volatile("s_wait_dscnt 0x0" ::: "memory");   // LDS RAW before reload
      // project gated tile onto states: K=16 real, pad K 16..31 with zero
      v16bf ga = hi ? cat8(z8, z8) : cat8(*(const v8bf*)&stage[wave][l16][0], *(const v8bf*)&stage[wave][l16][8]);
      accB[mi] = __builtin_amdgcn_wmma_f32_16x16x32_bf16(false, ga, false, wb,
                                               (short)0, accB[mi], false, false);
    }
  }

  // cross-wave reduction of partial B tiles
  #pragma unroll
  for (int mi = 0; mi < 4; ++mi)
    #pragma unroll
    for (int v = 0; v < 8; ++v)
      atomicAdd(&redB[(mi * 16 + v + (hi ? 8 : 0)) * DSZ + l16], accB[mi][v]);

  if (wave < 4) {  // C = (x+pos) @ W_C^T reuses the resident A strip; mi = wave
    v8f accC = {};
    const __bf16* wc = WCb + (size_t)l16 * DM + (hi ? 16 : 0);
    #pragma unroll 4
    for (int k0 = 0; k0 < DM; k0 += 32) {
      v16bf a  = cat8(*(const v8bf*)&Astrip[wave * 16 + l16][k0 + abase],
                      *(const v8bf*)&Astrip[wave * 16 + l16][k0 + abase + 8]);
      v16bf bc = cat8(*(const v8bf*)(wc + k0), *(const v8bf*)(wc + k0 + 8));
      accC = __builtin_amdgcn_wmma_f32_16x16x32_bf16(false, a, false, bc,
                                                     (short)0, accC, false, false);
    }
    #pragma unroll
    for (int v = 0; v < 8; ++v)
      redC[(wave * 16 + v + (hi ? 8 : 0)) * DSZ + l16] = accC[v];
  }
  __syncthreads();

  #pragma unroll
  for (int q = 0; q < 4; ++q) {   // 1024 (row, state) pairs, 4 per thread
    const int idx = tid + 256 * q;
    const int m = idx >> 4, n = idx & 15;
    const size_t r = (size_t)(row0 + m);
    Bm[r * DSZ + n] = redB[idx] + b_B[n];
    Cm[r * DSZ + n] = redC[idx] + b_C[n];
  }
}

// ---- 4) selective-scan recurrence -----------------------------------------
__global__ __launch_bounds__(256) void k_scan(const float* __restrict__ A_log,
                                              const float* __restrict__ Bm,
                                              const float* __restrict__ Cm,
                                              float* __restrict__ ym) {
  int idx = blockIdx.x * 256 + threadIdx.x;   // 8192 = 512 batches x 16 states
  int b = idx >> 4, n = idx & 15;
  float a = __expf(0.01f * __expf(A_log[n]));
  float h = 0.0f;
  const size_t base = (size_t)b * SEQLEN * DSZ + n;
  for (int s = 0; s < SEQLEN; ++s) {
    size_t o = base + (size_t)s * DSZ;
    h = a * h + Bm[o];
    ym[o] = h * Cm[o];
  }
}

// ---- 5) output projection: effectively K=16 (ref zero-pads), memory bound --
__global__ __launch_bounds__(256) void k_out(const float* __restrict__ ym,
                                             const float* __restrict__ Wout,
                                             const float* __restrict__ b_out,
                                             float* __restrict__ out) {
  __shared__ float ys[16];
  const int tid = threadIdx.x;
  const size_t row = blockIdx.x;
  if (tid < 16) ys[tid] = ym[row * DSZ + tid];
  __syncthreads();
  #pragma unroll
  for (int it = 0; it < 4; ++it) {
    int d = it * 256 + tid;
    const float* w = Wout + (size_t)d * DI;   // W_out[d, 0..15]
    float acc = b_out[d];
    #pragma unroll
    for (int i = 0; i < 16; ++i) acc += ys[i] * w[i];
    out[row * DM + d] = acc;
  }
}

// ---------------------------------------------------------------------------
extern "C" void kernel_launch(void* const* d_in, const int* in_sizes, int n_in,
                              void* d_out, int out_size, void* d_ws, size_t ws_size,
                              hipStream_t stream) {
  const float* x     = (const float*)d_in[0];
  const float* pos   = (const float*)d_in[1];
  const float* W_in  = (const float*)d_in[2];
  const float* b_in  = (const float*)d_in[3];
  const float* W_g   = (const float*)d_in[4];
  const float* b_g   = (const float*)d_in[5];
  const float* A_log = (const float*)d_in[6];
  const float* W_B   = (const float*)d_in[7];
  const float* b_B   = (const float*)d_in[8];
  const float* W_C   = (const float*)d_in[9];
  const float* b_C   = (const float*)d_in[10];
  const float* W_out = (const float*)d_in[11];
  const float* b_out = (const float*)d_in[12];
  float* out = (float*)d_out;

  char* ws = (char*)d_ws;
  size_t off = 0;
  auto take = [&](size_t bytes) -> char* {
    char* p = ws + off; off += (bytes + 255) & ~(size_t)255; return p;
  };
  __bf16* xb   = (__bf16*)take((size_t)NROWS * DM * 2);   // 104.9 MB
  __bf16* Winb = (__bf16*)take((size_t)DI * DM * 2);      //   4.2 MB
  __bf16* Wgb  = (__bf16*)take((size_t)DI * DM * 2);      //   4.2 MB
  __bf16* WBb  = (__bf16*)take((size_t)DSZ * DI * 2);
  __bf16* WCb  = (__bf16*)take((size_t)DSZ * DM * 2);
  float*  Bm   = (float*)take((size_t)NROWS * DSZ * 4);   //   3.3 MB
  float*  Cm   = (float*)take((size_t)NROWS * DSZ * 4);
  float*  ym   = (float*)take((size_t)NROWS * DSZ * 4);

  k_prep_x<<<(NROWS * DM) / 256, 256, 0, stream>>>(x, pos, xb);
  k_prep_w<<<(2 * DI * DM + DSZ * DI + DSZ * DM) / 256, 256, 0, stream>>>(
      W_in, W_g, W_B, W_C, Winb, Wgb, WBb, WCb);
  k_core<<<NROWS / MTILE, 256, 0, stream>>>(xb, Winb, Wgb, WBb, WCb,
                                            b_in, b_g, b_B, b_C, Bm, Cm);
  k_scan<<<(NBATCH * DSZ) / 256, 256, 0, stream>>>(A_log, Bm, Cm, ym);
  k_out<<<NROWS, 256, 0, stream>>>(ym, W_out, b_out, out);

  (void)in_sizes; (void)n_in; (void)out_size; (void)ws_size;
}
